// Attention_40175124086882
// MI455X (gfx1250) — matrix operand order repaired and checked by reading, not compiled, not run on hardware
//
#include <hip/hip_runtime.h>

typedef __attribute__((ext_vector_type(16))) _Float16 v16h;
typedef __attribute__((ext_vector_type(8)))  float    v8f;
typedef int v4i __attribute__((ext_vector_type(4)));

#define DIM  1024
#define NSEQ 2048
#define NH   16
#define DH   64
#define MROWS 4096   // 2 * 2048

// Async direct-to-LDS copy path (CDNA5 GLOBAL_LOAD_ASYNC_TO_LDS_*, ASYNCcnt)
#if defined(__gfx1250__) && __has_builtin(__builtin_amdgcn_global_load_async_to_lds_b128) && \
    __has_builtin(__builtin_amdgcn_s_wait_asynccnt)
#define ASYNC_CP 1
#else
#define ASYNC_CP 0
#endif

#if ASYNC_CP
static __device__ inline void async_cp16(const _Float16* g, _Float16* l) {
    __builtin_amdgcn_global_load_async_to_lds_b128(
        (__attribute__((address_space(1))) v4i*)g,
        (__attribute__((address_space(3))) v4i*)l, 0, 0);
}
#endif

// ---------------- WMMA fragment loaders (per ISA 7.12.2 layouts) ----------------
// A-matrix 16x32 f16: lane(0..15)=row M, hi=lane>>4 selects K-offset 8.
// VGPR v holds K = (v&3)*2 + (v>=4 ? 16 : 0) + hi*8, pairs contiguous.
static __device__ inline v16h frag_ld_a(const _Float16* base, int ld, int lane) {
    const int lr = lane & 15, hi = lane >> 4;
    const _Float16* r = base + lr * ld + hi * 8;
    union { v16h v; unsigned u[8]; } f;
#pragma unroll
    for (int i = 0; i < 8; ++i) {
        const int k = (i & 3) * 2 + ((i >> 2) * 16);
        f.u[i] = *(const unsigned*)(r + k);
    }
    return f.v;
}

// B-matrix 32x16 f16: lane(0..15)=column N, lanes 0-15 hold K=0..7,16..23, lanes 16-31 K=8..15,24..31.
static __device__ inline v16h frag_ld_b(const _Float16* base, int ld, int lane) {
    const int lr = lane & 15, hi = lane >> 4;
    const _Float16* r = base + lr * ld + hi * 8;
    union { v16h v; uint4 q[2]; } f;
    f.q[0] = *(const uint4*)(r);
    f.q[1] = *(const uint4*)(r + 16);
    return f.v;
}

static __device__ inline v8f wmma_f16(v16h a, v16h b, v8f c) {
    return __builtin_amdgcn_wmma_f32_16x16x32_f16(false, a, false, b, (short)0, c, false, false);
}

// ---------------- fp32 -> f16 convert ----------------
__global__ void conv_f32_f16(const float* __restrict__ src, _Float16* __restrict__ dst, int n) {
    int i = blockIdx.x * blockDim.x + threadIdx.x;
    const int stride = gridDim.x * blockDim.x;
    for (; i < n; i += stride) dst[i] = (_Float16)src[i];
}

// ---------------- Tiled GEMM: Y = A[M,1024] @ W[1024,1024]^T ----------------
// mode 0: f16 out, split-heads  [bh][n][64]
// mode 1: f16 out, transposed   [bh][64][n]   (for V)
// mode 2: f32 out, row-major    [m][1024]     (final projection)
__global__ __launch_bounds__(256)
void gemm_xwT(const _Float16* __restrict__ A, const _Float16* __restrict__ W,
              void* __restrict__ outp, int mode) {
    __shared__ _Float16 As[2][64 * 40];
    __shared__ _Float16 Bs[2][128 * 40];

    const int tid  = threadIdx.x;
    const int lane = tid & 31;
    const int wv   = tid >> 5;
    const int mBase = blockIdx.x * 64;
    const int nBase = blockIdx.y * 128;

    // per-thread staging coordinates (16B segments, coalesced)
    const int row = tid >> 2, seg = tid & 3;
    const _Float16* gA  = &A[(size_t)(mBase + row) * DIM + seg * 8];
    const _Float16* gB0 = &W[(size_t)(nBase + row) * DIM + seg * 8];
    const _Float16* gB1 = &W[(size_t)(nBase + row + 64) * DIM + seg * 8];
    const int lOff  = row * 40 + seg * 8;
    const int lOff1 = (row + 64) * 40 + seg * 8;

    v8f acc[4] = {};

    auto compute = [&](int cur) {
        const v16h bf = frag_ld_b(&Bs[cur][(wv * 16) * 40], 40, lane);
#pragma unroll
        for (int ms = 0; ms < 4; ++ms) {
            const v16h af = frag_ld_a(&As[cur][(ms * 16) * 40], 40, lane);
            acc[ms] = wmma_f16(af, bf, acc[ms]);
        }
    };

#if ASYNC_CP
    auto stage = [&](int kb, int bu) {
        async_cp16(gA + kb,  &As[bu][lOff]);
        async_cp16(gB0 + kb, &Bs[bu][lOff]);
        async_cp16(gB1 + kb, &Bs[bu][lOff1]);
    };
    stage(0, 0);
    for (int kb = 0, cur = 0; kb < DIM; kb += 32, cur ^= 1) {
        __builtin_amdgcn_s_wait_asynccnt(0);
        __syncthreads();                     // everyone's async fill of `cur` landed;
        if (kb + 32 < DIM) stage(kb + 32, cur ^ 1);  // and everyone finished reading cur^1
        compute(cur);
    }
#else
    for (int kb = 0; kb < DIM; kb += 32) {
        *(uint4*)&As[0][lOff]  = *(const uint4*)(gA + kb);
        *(uint4*)&Bs[0][lOff]  = *(const uint4*)(gB0 + kb);
        *(uint4*)&Bs[0][lOff1] = *(const uint4*)(gB1 + kb);
        __syncthreads();
        compute(0);
        __syncthreads();
    }
#endif

    const int lr = lane & 15, hi = lane >> 4;
    const int o = nBase + wv * 16 + lr;
#pragma unroll
    for (int ms = 0; ms < 4; ++ms) {
#pragma unroll
        for (int r = 0; r < 8; ++r) {
            const int m = mBase + ms * 16 + r + hi * 8;
            const float val = acc[ms][r];
            if (mode == 2) {
                ((float*)outp)[(size_t)m * DIM + o] = val;
            } else {
                const int b = m >> 11, nn = m & 2047;
                const int h = o >> 6,  dd = o & 63;
                if (mode == 0)
                    ((_Float16*)outp)[(((size_t)(b * NH + h)) * NSEQ + nn) * DH + dd] = (_Float16)val;
                else
                    ((_Float16*)outp)[(((size_t)(b * NH + h)) * DH + dd) * NSEQ + nn] = (_Float16)val;
            }
        }
    }
}

// ---------------- Flash attention (causal), one wave = 16 query rows ----------------
__global__ __launch_bounds__(128)
void flash_attn(const _Float16* __restrict__ Q, const _Float16* __restrict__ K,
                const _Float16* __restrict__ Vt, _Float16* __restrict__ Out) {
    __shared__ _Float16 Pb[4][16 * 40];   // per-wave P scratch (D-layout -> A-layout)

    const int lane = threadIdx.x & 31;
    const int wv   = threadIdx.x >> 5;
    const int lr   = lane & 15, hi = lane >> 4;
    const int bh   = blockIdx.y;
    const int b    = bh >> 4, h = bh & 15;
    const int mBase = blockIdx.x * 64 + wv * 16;

    const _Float16* Qh = Q  + (size_t)bh * NSEQ * DH;
    const _Float16* Kh = K  + (size_t)bh * NSEQ * DH;
    const _Float16* Vh = Vt + (size_t)bh * DH * NSEQ;   // [64][2048]

    const v16h aQ0 = frag_ld_a(Qh + (size_t)mBase * DH,      DH, lane);
    const v16h aQ1 = frag_ld_a(Qh + (size_t)mBase * DH + 32, DH, lane);

    v8f o[4] = {};
    float m_i[8], l_i[8];
#pragma unroll
    for (int r = 0; r < 8; ++r) { m_i[r] = -1e30f; l_i[r] = 0.f; }

    const float scale = 0.125f;             // 1/sqrt(64)
    const int jEnd = mBase + 16;            // causal limit for this wave's rows

    for (int jb = 0; jb < jEnd; jb += 32) {
        // S = Q @ K^T  (k-dim = head dim, split 0..31 / 32..63)
        const v16h bk00 = frag_ld_b(Kh + (size_t)jb * DH,             DH, lane);
        const v16h bk01 = frag_ld_b(Kh + (size_t)jb * DH + 32,        DH, lane);
        const v16h bk10 = frag_ld_b(Kh + (size_t)(jb + 16) * DH,      DH, lane);
        const v16h bk11 = frag_ld_b(Kh + (size_t)(jb + 16) * DH + 32, DH, lane);
        v8f s0 = {}, s1 = {};
        s0 = wmma_f16(aQ0, bk00, s0);
        s0 = wmma_f16(aQ1, bk01, s0);
        s1 = wmma_f16(aQ0, bk10, s1);
        s1 = wmma_f16(aQ1, bk11, s1);

        // online softmax per row (row = mBase + r + hi*8, cols across 16-lane group)
#pragma unroll
        for (int r = 0; r < 8; ++r) {
            const int row = mBase + r + hi * 8;
            s0[r] = (jb + lr      <= row) ? s0[r] * scale : -1e30f;
            s1[r] = (jb + 16 + lr <= row) ? s1[r] * scale : -1e30f;
            float t = fmaxf(s0[r], s1[r]);
            t = fmaxf(t, __shfl_xor(t, 1, 32));
            t = fmaxf(t, __shfl_xor(t, 2, 32));
            t = fmaxf(t, __shfl_xor(t, 4, 32));
            t = fmaxf(t, __shfl_xor(t, 8, 32));
            const float mnew = fmaxf(m_i[r], t);
            const float corr = __expf(m_i[r] - mnew);
            const float p0 = __expf(s0[r] - mnew);
            const float p1 = __expf(s1[r] - mnew);
            float rs = p0 + p1;
            rs += __shfl_xor(rs, 1, 32);
            rs += __shfl_xor(rs, 2, 32);
            rs += __shfl_xor(rs, 4, 32);
            rs += __shfl_xor(rs, 8, 32);
            l_i[r] = l_i[r] * corr + rs;
            m_i[r] = mnew;
#pragma unroll
            for (int ns = 0; ns < 4; ++ns) o[ns][r] *= corr;
            Pb[wv][(r + hi * 8) * 40 + lr]      = (_Float16)p0;
            Pb[wv][(r + hi * 8) * 40 + 16 + lr] = (_Float16)p1;
        }

        // O += P @ V   (P: 16x32 A-frag via LDS; V^T rows give contiguous B-frags)
        const v16h aP = frag_ld_a(&Pb[wv][0], 40, lane);
#pragma unroll
        for (int ns = 0; ns < 4; ++ns) {
            const v16h bv = frag_ld_b(Vh + (size_t)(ns * 16) * NSEQ + jb, NSEQ, lane);
            o[ns] = wmma_f16(aP, bv, o[ns]);
        }
    }

    // normalize and write out as [b][n][h*64+dd]
    _Float16* Ob = Out + (size_t)b * NSEQ * DIM + h * DH;
#pragma unroll
    for (int r = 0; r < 8; ++r) {
        const float inv = 1.0f / l_i[r];
        const int row = mBase + r + hi * 8;
#pragma unroll
        for (int ns = 0; ns < 4; ++ns)
            Ob[(size_t)row * DIM + ns * 16 + lr] = (_Float16)(o[ns][r] * inv);
    }
}

// ---------------- launch ----------------
extern "C" void kernel_launch(void* const* d_in, const int* in_sizes, int n_in,
                              void* d_out, int out_size, void* d_ws, size_t ws_size,
                              hipStream_t stream) {
    (void)in_sizes; (void)n_in; (void)out_size; (void)ws_size;
    const float* x  = (const float*)d_in[0];
    const float* Wq = (const float*)d_in[1];
    const float* Wk = (const float*)d_in[2];
    const float* Wv = (const float*)d_in[3];
    const float* Wo = (const float*)d_in[4];

    char* ws = (char*)d_ws;
    const size_t MB = 1u << 20;
    _Float16* x16  = (_Float16*)(ws + 0);          // 8 MB
    _Float16* wq16 = (_Float16*)(ws + 8  * MB);    // 2 MB each
    _Float16* wk16 = (_Float16*)(ws + 10 * MB);
    _Float16* wv16 = (_Float16*)(ws + 12 * MB);
    _Float16* wo16 = (_Float16*)(ws + 14 * MB);
    _Float16* q16  = (_Float16*)(ws + 16 * MB);    // [32][2048][64]
    _Float16* k16  = (_Float16*)(ws + 24 * MB);    // [32][2048][64]
    _Float16* vT16 = (_Float16*)(ws + 32 * MB);    // [32][64][2048]
    _Float16* ao16 = (_Float16*)(ws + 40 * MB);    // [4096][1024]

    conv_f32_f16<<<2048, 256, 0, stream>>>(x,  x16,  MROWS * DIM);
    conv_f32_f16<<<1024, 256, 0, stream>>>(Wq, wq16, DIM * DIM);
    conv_f32_f16<<<1024, 256, 0, stream>>>(Wk, wk16, DIM * DIM);
    conv_f32_f16<<<1024, 256, 0, stream>>>(Wv, wv16, DIM * DIM);
    conv_f32_f16<<<1024, 256, 0, stream>>>(Wo, wo16, DIM * DIM);

    dim3 ggrid(MROWS / 64, DIM / 128);
    gemm_xwT<<<ggrid, 256, 0, stream>>>(x16, wq16, (void*)q16,  0);
    gemm_xwT<<<ggrid, 256, 0, stream>>>(x16, wk16, (void*)k16,  0);
    gemm_xwT<<<ggrid, 256, 0, stream>>>(x16, wv16, (void*)vT16, 1);

    flash_attn<<<dim3(NSEQ / 64, 32), 128, 0, stream>>>(q16, k16, vT16, ao16);

    gemm_xwT<<<ggrid, 256, 0, stream>>>(ao16, wo16, d_out, 2);
}
